// Qwen3MoeSparseMoeBlockParaS_41188736369343
// MI455X (gfx1250) — matrix operand order repaired and checked by reading, not compiled, not run on hardware
//
#include <hip/hip_runtime.h>

constexpr int kT  = 512;
constexpr int kD  = 2048;
constexpr int kFF = 768;
constexpr int kE  = 128;
constexpr int kK  = 8;

typedef __attribute__((ext_vector_type(16))) __bf16 v16bf;
typedef __attribute__((ext_vector_type(8)))  float  v8f;
typedef __attribute__((ext_vector_type(2)))  float  v2f;
typedef __attribute__((ext_vector_type(2)))  __bf16 v2bf;

union FragBF { v16bf v; unsigned int u[8]; };
union FragF  { v8f   v; float        f[8]; };

// float2 -> packed bf16x2 (lowers to v_cvt_pk_bf16_f32).
__device__ __forceinline__ unsigned int pack2bf(float lo, float hi) {
  union { v2bf v; unsigned int u; } c;
  v2f in; in.x = lo; in.y = hi;
  c.v = __builtin_convertvector(in, v2bf);
  return c.u;
}

__device__ __forceinline__ unsigned short f32_to_bf16(float x) {
  union { v2bf v; unsigned short s[2]; } c;
  v2f in; in.x = x; in.y = 0.f;
  c.v = __builtin_convertvector(in, v2bf);
  return c.s[0];
}

// fast silu(g)*u: bf16 output tolerates approx rcp (single v_rcp_f32)
__device__ __forceinline__ float swiglu(float g, float u) {
  return g * __builtin_amdgcn_rcpf(1.f + __expf(-g)) * u;
}

// ---------------- init: zero output + per-expert counters ----------------
__global__ void init_kernel(float* __restrict__ out, int* __restrict__ counts,
                            int* __restrict__ counts2) {
  int i = blockIdx.x * blockDim.x + threadIdx.x;
  if (i < kT * kD) out[i] = 0.f;
  if (i < kE) { counts[i] = 0; counts2[i] = 0; }
}

// ---------------- convert x (fp32 -> bf16) once; 4MB read, 2MB write ----------------
__global__ void cvt_x_kernel(const float* __restrict__ x, unsigned int* __restrict__ xb) {
  int i = blockIdx.x * blockDim.x + threadIdx.x;   // over pairs
  if (i < kT * kD / 2) {
    const float2 p = ((const float2*)x)[i];
    xb[i] = pack2bf(p.x, p.y);
  }
}

// ---------------- router: logits -> softmax -> top-8 -> renorm ----------------
__global__ void __launch_bounds__(kE)
router_kernel(const float* __restrict__ x, const float* __restrict__ wgate,
              float* __restrict__ combine_w, int* __restrict__ topk_idx) {
  __shared__ float xs[kD];
  __shared__ float prob[kE];
  const int t = blockIdx.x;
  for (int i = threadIdx.x; i < kD; i += blockDim.x) xs[i] = x[(size_t)t * kD + i];
  __syncthreads();
  const int e = threadIdx.x;
  float acc = 0.f;
  for (int k = 0; k < kD; ++k) acc += xs[k] * wgate[(size_t)k * kE + e];
  prob[e] = acc;
  __syncthreads();
  if (threadIdx.x == 0) {
    float mx = prob[0];
    for (int i = 1; i < kE; ++i) mx = fmaxf(mx, prob[i]);
    float s = 0.f;
    for (int i = 0; i < kE; ++i) { float p = __expf(prob[i] - mx); prob[i] = p; s += p; }
    const float inv = 1.f / s;
    int idxs[kK]; float wsk[kK]; float wsum = 0.f;
    for (int k = 0; k < kK; ++k) {
      int bi = 0; float bv = -1.f;
      for (int i = 0; i < kE; ++i) if (prob[i] > bv) { bv = prob[i]; bi = i; }
      prob[bi] = -2.f;
      idxs[k] = bi; wsk[k] = bv * inv; wsum += bv * inv;
    }
    const float rn = 1.f / wsum;
    for (int k = 0; k < kK; ++k) {
      combine_w[t * kK + k] = wsk[k] * rn;
      topk_idx[t * kK + k]  = idxs[k];
    }
  }
}

// ---------------- dispatch ----------------
__global__ void count_kernel(const int* __restrict__ topk_idx, int* __restrict__ counts) {
  int i = blockIdx.x * blockDim.x + threadIdx.x;
  if (i < kT * kK) atomicAdd(&counts[topk_idx[i]], 1);
}

__global__ void scan_kernel(const int* __restrict__ counts, int* __restrict__ offsets) {
  if (blockIdx.x == 0 && threadIdx.x == 0) {
    int s = 0;
    for (int e = 0; e < kE; ++e) { offsets[e] = s; s += counts[e]; }
  }
}

__global__ void fill_kernel(const int* __restrict__ topk_idx, const float* __restrict__ combine_w,
                            const int* __restrict__ offsets, int* __restrict__ counts2,
                            int* __restrict__ tok_of, float* __restrict__ w_of) {
  int i = blockIdx.x * blockDim.x + threadIdx.x;
  if (i < kT * kK) {
    const int e = topk_idx[i];
    const int slot = atomicAdd(&counts2[e], 1);
    const int row = offsets[e] + slot;
    tok_of[row] = i / kK;
    w_of[row]   = combine_w[i];
  }
}

__device__ __forceinline__ void load_a_frag(FragBF& a, const unsigned short* row,
                                            int k0, bool hi) {
  const uint4 A1 = *(const uint4*)(row + k0 + (hi ? 8 : 0));
  const uint4 A2 = *(const uint4*)(row + k0 + 16 + (hi ? 8 : 0));
  a.u[0] = A1.x; a.u[1] = A1.y; a.u[2] = A1.z; a.u[3] = A1.w;
  a.u[4] = A2.x; a.u[5] = A2.y; a.u[6] = A2.z; a.u[7] = A2.w;
}

// K-loop for gate/up, TWO hoisted out of the loop (cnt is WG-uniform).
template <bool TWO>
__device__ __forceinline__ void gateup_mm(const unsigned short* __restrict__ x0,
                                          const unsigned short* __restrict__ x1,
                                          const float* __restrict__ wgp_e,
                                          const float* __restrict__ wup_e,
                                          int col, bool hi,
                                          FragF& ag0, FragF& au0, FragF& ag1, FragF& au1) {
  for (int k0 = 0; k0 < kD; k0 += 32) {
    FragBF a0, a1, bg, bu;
    load_a_frag(a0, x0, k0, hi);
    if (TWO) load_a_frag(a1, x1, k0, hi);
#pragma unroll
    for (int r = 0; r < 8; ++r) {
      const int kb = k0 + 2 * r + (r >= 4 ? 8 : 0) + (hi ? 8 : 0);
      const size_t o0 = (size_t)kb * kFF + col;
      bg.u[r] = pack2bf(wgp_e[o0], wgp_e[o0 + kFF]);
      bu.u[r] = pack2bf(wup_e[o0], wup_e[o0 + kFF]);
    }
    if (k0 + 32 < kD) {
      __builtin_prefetch(&wgp_e[(size_t)(k0 + 32) * kFF + col], 0, 1);
      __builtin_prefetch(&wup_e[(size_t)(k0 + 32) * kFF + col], 0, 1);
    }
    ag0.v = __builtin_amdgcn_wmma_f32_16x16x32_bf16(false, a0.v, false, bg.v, (short)0, ag0.v, false, false);
    au0.v = __builtin_amdgcn_wmma_f32_16x16x32_bf16(false, a0.v, false, bu.v, (short)0, au0.v, false, false);
    if (TWO) {
      ag1.v = __builtin_amdgcn_wmma_f32_16x16x32_bf16(false, a1.v, false, bg.v, (short)0, ag1.v, false, false);
      au1.v = __builtin_amdgcn_wmma_f32_16x16x32_bf16(false, a1.v, false, bu.v, (short)0, au1.v, false, false);
    }
  }
}

// ---------------- expert gate/up GEMM + SwiGLU -> h (bf16) ----------------
// grid: (kFF/128, kE), 256 threads = 8 waves; wave owns a 16-wide N strip.
// M-block = 32: each streamed/converted B fragment feeds 2 WMMAs.
__global__ void __launch_bounds__(256)
gateup_kernel(const unsigned short* __restrict__ xb, const float* __restrict__ wgp,
              const float* __restrict__ wup, const int* __restrict__ counts,
              const int* __restrict__ offsets, const int* __restrict__ tok_of,
              unsigned short* __restrict__ h_ws) {
  const int e = blockIdx.y;
  const int cnt = counts[e];
  if (cnt == 0) return;
  const int base = offsets[e];
  const int lane = threadIdx.x & 31;
  const int wv   = threadIdx.x >> 5;
  const bool hi  = lane >= 16;
  const int lm   = lane & 15;
  const int nblk = blockIdx.x * 128 + wv * 16;
  const float* wgp_e = wgp + (size_t)e * kD * kFF;
  const float* wup_e = wup + (size_t)e * kD * kFF;

  for (int m0 = 0; m0 < cnt; m0 += 32) {
    const bool two = (m0 + 16 < cnt);                 // WG-uniform
    int rm0 = m0 + lm;      if (rm0 >= cnt) rm0 = cnt - 1;
    int rm1 = m0 + 16 + lm; if (rm1 >= cnt) rm1 = cnt - 1;
    const unsigned short* x0 = xb + (size_t)tok_of[base + rm0] * kD;
    const unsigned short* x1 = xb + (size_t)tok_of[base + rm1] * kD;

    FragF ag0, au0, ag1, au1;
#pragma unroll
    for (int i = 0; i < 8; ++i) { ag0.f[i] = 0.f; au0.f[i] = 0.f; ag1.f[i] = 0.f; au1.f[i] = 0.f; }

    if (two) gateup_mm<true >(x0, x1, wgp_e, wup_e, nblk + lm, hi, ag0, au0, ag1, au1);
    else     gateup_mm<false>(x0, x1, wgp_e, wup_e, nblk + lm, hi, ag0, au0, ag1, au1);

    // C layout: VGPR i -> (M=i, N=lane&15) lanes 0-15; (M=8+i) lanes 16-31
#pragma unroll
    for (int i = 0; i < 8; ++i) {
      const int m = i + (hi ? 8 : 0);
      if (m0 + m < cnt)
        h_ws[(size_t)(base + m0 + m) * kFF + nblk + lm] = f32_to_bf16(swiglu(ag0.f[i], au0.f[i]));
      if (m0 + 16 + m < cnt)
        h_ws[(size_t)(base + m0 + 16 + m) * kFF + nblk + lm] = f32_to_bf16(swiglu(ag1.f[i], au1.f[i]));
    }
  }
}

// K-loop for down projection, TWO hoisted.
template <bool TWO>
__device__ __forceinline__ void down_mm(const unsigned short* __restrict__ h0,
                                        const unsigned short* __restrict__ h1,
                                        const float* __restrict__ wd_e,
                                        int col, bool hi, FragF& ac0, FragF& ac1) {
  for (int k0 = 0; k0 < kFF; k0 += 32) {
    FragBF a0, a1, b;
    load_a_frag(a0, h0, k0, hi);
    if (TWO) load_a_frag(a1, h1, k0, hi);
#pragma unroll
    for (int r = 0; r < 8; ++r) {
      const int kb = k0 + 2 * r + (r >= 4 ? 8 : 0) + (hi ? 8 : 0);
      const size_t o0 = (size_t)kb * kD + col;
      b.u[r] = pack2bf(wd_e[o0], wd_e[o0 + kD]);
    }
    if (k0 + 32 < kFF) __builtin_prefetch(&wd_e[(size_t)(k0 + 32) * kD + col], 0, 1);
    ac0.v = __builtin_amdgcn_wmma_f32_16x16x32_bf16(false, a0.v, false, b.v, (short)0, ac0.v, false, false);
    if (TWO)
      ac1.v = __builtin_amdgcn_wmma_f32_16x16x32_bf16(false, a1.v, false, b.v, (short)0, ac1.v, false, false);
  }
}

// ---------------- down GEMM + weighted combine (atomic f32) ----------------
// grid: (kD/128, kE), 256 threads = 8 waves; M-block = 32.
__global__ void __launch_bounds__(256)
down_kernel(const float* __restrict__ wdn, const unsigned short* __restrict__ h_ws,
            const int* __restrict__ counts, const int* __restrict__ offsets,
            const int* __restrict__ tok_of, const float* __restrict__ w_of,
            float* __restrict__ out) {
  const int e = blockIdx.y;
  const int cnt = counts[e];
  if (cnt == 0) return;
  const int base = offsets[e];
  const int lane = threadIdx.x & 31;
  const int wv   = threadIdx.x >> 5;
  const bool hi  = lane >= 16;
  const int lm   = lane & 15;
  const int nblk = blockIdx.x * 128 + wv * 16;
  const float* wd_e = wdn + (size_t)e * kFF * kD;

  for (int m0 = 0; m0 < cnt; m0 += 32) {
    const bool two = (m0 + 16 < cnt);                 // WG-uniform
    int rm0 = m0 + lm;      if (rm0 >= cnt) rm0 = cnt - 1;
    int rm1 = m0 + 16 + lm; if (rm1 >= cnt) rm1 = cnt - 1;
    const unsigned short* h0 = h_ws + (size_t)(base + rm0) * kFF;
    const unsigned short* h1 = h_ws + (size_t)(base + rm1) * kFF;

    FragF ac0, ac1;
#pragma unroll
    for (int i = 0; i < 8; ++i) { ac0.f[i] = 0.f; ac1.f[i] = 0.f; }

    if (two) down_mm<true >(h0, h1, wd_e, nblk + lm, hi, ac0, ac1);
    else     down_mm<false>(h0, h1, wd_e, nblk + lm, hi, ac0, ac1);

#pragma unroll
    for (int i = 0; i < 8; ++i) {
      const int m = i + (hi ? 8 : 0);
      if (m0 + m < cnt) {
        const int row = base + m0 + m;
        atomicAdd(out + (size_t)tok_of[row] * kD + nblk + lm, w_of[row] * ac0.f[i]);
      }
      if (m0 + 16 + m < cnt) {
        const int row = base + m0 + 16 + m;
        atomicAdd(out + (size_t)tok_of[row] * kD + nblk + lm, w_of[row] * ac1.f[i]);
      }
    }
  }
}

// ---------------- launch ----------------
extern "C" void kernel_launch(void* const* d_in, const int* in_sizes, int n_in,
                              void* d_out, int out_size, void* d_ws, size_t ws_size,
                              hipStream_t stream) {
  const float* x   = (const float*)d_in[0];
  const float* wg  = (const float*)d_in[1];
  const float* wgp = (const float*)d_in[2];
  const float* wup = (const float*)d_in[3];
  const float* wdn = (const float*)d_in[4];
  float* out = (float*)d_out;

  char* ws = (char*)d_ws;
  float* combine_w = (float*)ws;  ws += (size_t)kT * kK * sizeof(float);
  int*   topk_idx  = (int*)ws;    ws += (size_t)kT * kK * sizeof(int);
  int*   counts    = (int*)ws;    ws += (size_t)kE * sizeof(int);
  int*   counts2   = (int*)ws;    ws += (size_t)kE * sizeof(int);
  int*   offsets   = (int*)ws;    ws += (size_t)kE * sizeof(int);
  int*   tok_of    = (int*)ws;    ws += (size_t)kT * kK * sizeof(int);
  float* w_of      = (float*)ws;  ws += (size_t)kT * kK * sizeof(float);
  ws = (char*)(((unsigned long long)ws + 15ull) & ~15ull);
  unsigned short* xb = (unsigned short*)ws;  ws += (size_t)kT * kD * sizeof(unsigned short); // 2MB
  unsigned short* h_ws = (unsigned short*)ws;   // kT*kK rows x kFF bf16 (~6.3 MB)

  init_kernel<<<(kT * kD + 255) / 256, 256, 0, stream>>>(out, counts, counts2);
  cvt_x_kernel<<<(kT * kD / 2 + 255) / 256, 256, 0, stream>>>(x, (unsigned int*)xb);
  router_kernel<<<kT, kE, 0, stream>>>(x, wg, combine_w, topk_idx);
  count_kernel<<<(kT * kK + 255) / 256, 256, 0, stream>>>(topk_idx, counts);
  scan_kernel<<<1, 1, 0, stream>>>(counts, offsets);
  fill_kernel<<<(kT * kK + 255) / 256, 256, 0, stream>>>(topk_idx, combine_w, offsets,
                                                         counts2, tok_of, w_of);
  gateup_kernel<<<dim3(kFF / 128, kE), 256, 0, stream>>>(xb, wgp, wup, counts, offsets,
                                                         tok_of, h_ws);
  down_kernel<<<dim3(kD / 128, kE), 256, 0, stream>>>(wdn, h_ws, counts, offsets,
                                                      tok_of, w_of, out);
}
